// CANN_72395968741836
// MI455X (gfx1250) — matrix operand order repaired and checked by reading, not compiled, not run on hardware
//
#include <hip/hip_runtime.h>

#define N_ASSETS 8192
#define DIM 512
#define BR 64         // query rows per block
#define BC 128        // keys per iteration
#define QSTRIDE 520   // 512 bf16 padded: 260 dwords (=4 mod 64) -> conflict-free b128
#define SSTRIDE 132   // 128 f32 padded
#define PSTRIDE 136   // 128 bf16 padded (68 dwords)

typedef __attribute__((ext_vector_type(16))) __bf16 v16bf;
typedef __attribute__((ext_vector_type(8)))  __bf16 v8bf;
typedef __attribute__((ext_vector_type(4)))  __bf16 v4bf;
typedef __attribute__((ext_vector_type(8)))  float  v8f;

__device__ __forceinline__ v8f wmma_bf16(v16bf a, v16bf b, v8f c) {
  // D = A(16x32 bf16) * B(32x16 bf16) + C(16x16 f32)
  return __builtin_amdgcn_wmma_f32_16x16x32_bf16(false, a, false, b, (short)0, c, false, false);
}

// B fragment: lane holds 16 contiguous bf16 along K (32B aligned)
__device__ __forceinline__ v16bf load_b16(const __bf16* p) {
  return *(const v16bf*)p;
}

// A fragment: elements 0..15 at k+half*16 (contiguous)
__device__ __forceinline__ v16bf make_a(const __bf16* p) {
  v8bf lo = *(const v8bf*)(p);
  v8bf hi = *(const v8bf*)(p + 8);
  v16bf a;
#pragma unroll
  for (int i = 0; i < 8; ++i) { a[i] = lo[i]; a[i + 8] = hi[i]; }
  return a;
}

// ---------------------------------------------------------------------------
// Pre-pass: convert z (N x D) and Wq/Wk/Wv (D x D) to bf16 once, so the
// projection GEMM inner loop is pure b128-load + WMMA (no f32->bf16 chains).
// ---------------------------------------------------------------------------
__global__ __launch_bounds__(256) void CANN_cvt_kernel(
    const float* __restrict__ z,  const float* __restrict__ Wq,
    const float* __restrict__ Wk, const float* __restrict__ Wv,
    __bf16* __restrict__ zb, __bf16* __restrict__ Wb)
{
  const int NZ = N_ASSETS * DIM;   // 4194304
  const int NW = DIM * DIM;        // 262144
  const int i4 = (blockIdx.x * 256 + threadIdx.x) * 4;
  const float* src;
  __bf16* dst;
  if (i4 < NZ) {
    src = z + i4;
    dst = zb + i4;
  } else {
    const int j = i4 - NZ;                 // 0 .. 3*NW-1
    src = (j < NW) ? (Wq + j) : (j < 2 * NW) ? (Wk + (j - NW)) : (Wv + (j - 2 * NW));
    dst = Wb + j;
  }
  float4 x = *(const float4*)src;
  v4bf o;
  o[0] = (__bf16)x.x; o[1] = (__bf16)x.y; o[2] = (__bf16)x.z; o[3] = (__bf16)x.w;
  *(v4bf*)dst = o;
}

// ---------------------------------------------------------------------------
// Projection: out = z @ W^T + b for Q, K (row-major bf16) and V (transposed,
// D x N bf16 so P@V B-fragments are contiguous). One 16x16 tile per wave.
// ---------------------------------------------------------------------------
__global__ __launch_bounds__(256) void CANN_proj_kernel(
    const __bf16* __restrict__ zb, const __bf16* __restrict__ Wb,
    const float* __restrict__ bq, const float* __restrict__ bk,
    const float* __restrict__ bv,
    __bf16* __restrict__ Qb, __bf16* __restrict__ Kb, __bf16* __restrict__ Vt)
{
  const int proj = blockIdx.y;
  const float* bias = (proj == 0) ? bq : (proj == 1) ? bk : bv;
  const __bf16* W = Wb + (size_t)proj * DIM * DIM;

  const int tid  = threadIdx.x;
  const int lane = tid & 31;
  const int wave = tid >> 5;
  const int tile = blockIdx.x * 8 + wave;
  const int m0   = (tile >> 5) << 4;
  const int j0   = (tile & 31) << 4;
  const int col  = lane & 15;
  const int half = lane >> 4;

  const __bf16* arow = zb + (size_t)(m0 + col) * DIM + half * 16;  // A row, elems at +0 / +8
  const __bf16* brow = W  + (size_t)(j0 + col) * DIM + half * 16;  // B row (W^T access)

  v8f c = {};
#pragma unroll 4
  for (int kb = 0; kb < DIM; kb += 32) {
    v16bf a = make_a(arow + kb);
    v16bf b = load_b16(brow + kb);
    c = wmma_bf16(a, b, c);
  }

  const float bias_v = bias[j0 + col];
  const int rbase = half * 8;
  if (proj < 2) {
    __bf16* out = (proj == 0) ? Qb : Kb;
#pragma unroll
    for (int r = 0; r < 8; ++r)
      out[(size_t)(m0 + rbase + r) * DIM + j0 + col] = (__bf16)(c[r] + bias_v);
  } else {
    v8bf p;
#pragma unroll
    for (int r = 0; r < 8; ++r) p[r] = (__bf16)(c[r] + bias_v);
    *(v8bf*)(Vt + (size_t)(j0 + col) * N_ASSETS + m0 + rbase) = p;  // contiguous b128 store
  }
}

// ---------------------------------------------------------------------------
// Flash attention, Br=64 rows per 256-thread block (8 waves), Bc=128 keys/iter.
// S phase: wave w owns key-tile w; one K B-fragment feeds 4 independent WMMAs
// (the 4 Q row-tiles) -> K read exactly once per block.
// PV phase: wave w owns d-cols [w*64,+64); one V B-fragment feeds 4 WMMAs
// (the 4 row-tiles) -> V read exactly once per block.
// O: 16 f32 C-fragments (128 VGPRs) per wave.
// ---------------------------------------------------------------------------
__global__ __launch_bounds__(256, 1) void CANN_attn_kernel(
    const __bf16* __restrict__ Qb, const __bf16* __restrict__ Kb,
    const __bf16* __restrict__ Vt, float* __restrict__ out)
{
  __shared__ __align__(16) __bf16 sQ[BR * QSTRIDE];
  __shared__ float  sS[BR * SSTRIDE];
  __shared__ __align__(16) __bf16 sP[BR * PSTRIDE];
  __shared__ float  pmax[4 * BR];
  __shared__ float  psum[4 * BR];
  __shared__ float  sm[BR], sl[BR], salpha[BR];

  const int tid   = threadIdx.x;
  const int lane  = tid & 31;
  const int wave  = tid >> 5;
  const int col   = lane & 15;
  const int half  = lane >> 4;
  const int rbase = half * 8;
  const int m0    = blockIdx.x * BR;

  // stage the Q tile (64 x 512 bf16) into padded LDS
  {
    const unsigned int* src = (const unsigned int*)(Qb + (size_t)m0 * DIM);
    unsigned int* dst = (unsigned int*)sQ;
    for (int idx = tid; idx < BR * (DIM / 2); idx += 256) {
      const int row = idx >> 8;          // DIM/2 = 256 dwords/row
      const int cd  = idx & 255;
      dst[row * (QSTRIDE / 2) + cd] = src[row * (DIM / 2) + cd];
    }
  }
  if (tid < BR) { sm[tid] = -1e30f; sl[tid] = 0.0f; }
  __syncthreads();

  v8f acc[16];
#pragma unroll
  for (int i = 0; i < 16; ++i) acc[i] = (v8f){};
  const float scale = 0.044194173824159216f; // 512^-0.5

  for (int n0 = 0; n0 < N_ASSETS; n0 += BC) {
    // ---- S = scale * Q @ K^T : wave w -> key cols [w*16,+16), all 4 row tiles
    {
      const __bf16* krow = Kb + (size_t)(n0 + wave * 16 + col) * DIM + half * 16;
      v8f s[4] = {{}, {}, {}, {}};
#pragma unroll 4
      for (int kb = 0; kb < DIM; kb += 32) {
        v16bf b = load_b16(krow + kb);
#pragma unroll
        for (int rt = 0; rt < 4; ++rt) {
          v16bf a = make_a(sQ + (size_t)(rt * 16 + col) * QSTRIDE + kb + half * 16);
          s[rt] = wmma_bf16(a, b, s[rt]);
        }
      }
#pragma unroll
      for (int rt = 0; rt < 4; ++rt)
#pragma unroll
        for (int r = 0; r < 8; ++r)
          sS[(rt * 16 + rbase + r) * SSTRIDE + wave * 16 + col] = s[rt][r] * scale;
    }
    __syncthreads();

    // ---- online softmax stats: 4 threads per row, 32 cols each
    const int row   = tid & 63;
    const int chunk = tid >> 6;   // 0..3
    {
      float mx = -1e30f;
#pragma unroll
      for (int q = 0; q < 32; ++q)
        mx = fmaxf(mx, sS[row * SSTRIDE + chunk * 32 + q]);
      pmax[chunk * BR + row] = mx;
    }
    __syncthreads();
    if (tid < BR) {
      float mm = sm[tid];
#pragma unroll
      for (int q = 0; q < 4; ++q) mm = fmaxf(mm, pmax[q * BR + tid]);
      salpha[tid] = __expf(sm[tid] - mm);
      sm[tid] = mm;
    }
    __syncthreads();
    {
      const float mnew = sm[row];
      float lsum = 0.0f;
#pragma unroll
      for (int q = 0; q < 32; ++q) {
        const float p = __expf(sS[row * SSTRIDE + chunk * 32 + q] - mnew);
        sP[row * PSTRIDE + chunk * 32 + q] = (__bf16)p;
        lsum += p;
      }
      psum[chunk * BR + row] = lsum;
    }
    __syncthreads();
    if (tid < BR) {
      float ls = 0.0f;
#pragma unroll
      for (int q = 0; q < 4; ++q) ls += psum[q * BR + tid];
      sl[tid] = sl[tid] * salpha[tid] + ls;
    }
    __syncthreads();

    // ---- O = O*alpha + P @ V : wave w -> d-cols [w*64,+64), all 4 row tiles
    {
#pragma unroll
      for (int rt = 0; rt < 4; ++rt) {
        float al[8];
#pragma unroll
        for (int r = 0; r < 8; ++r) al[r] = salpha[rt * 16 + rbase + r];
#pragma unroll
        for (int jt = 0; jt < 4; ++jt)
#pragma unroll
          for (int r = 0; r < 8; ++r) acc[rt * 4 + jt][r] *= al[r];
      }
#pragma unroll
      for (int jt = 0; jt < 4; ++jt) {
        const __bf16* vrow =
            Vt + (size_t)(wave * 64 + jt * 16 + col) * N_ASSETS + n0 + half * 16;
#pragma unroll
        for (int kb = 0; kb < BC; kb += 32) {
          v16bf b = load_b16(vrow + kb);
#pragma unroll
          for (int rt = 0; rt < 4; ++rt) {
            v16bf a = make_a(sP + (size_t)(rt * 16 + col) * PSTRIDE + kb + half * 16);
            acc[rt * 4 + jt] = wmma_bf16(a, b, acc[rt * 4 + jt]);
          }
        }
      }
    }
    __syncthreads();
  }

  // ---- finalize: divide by l, write fp32 output
#pragma unroll
  for (int rt = 0; rt < 4; ++rt) {
    float linv[8];
#pragma unroll
    for (int r = 0; r < 8; ++r) linv[r] = 1.0f / sl[rt * 16 + rbase + r];
#pragma unroll
    for (int jt = 0; jt < 4; ++jt) {
      const int oc = wave * 64 + jt * 16 + col;
#pragma unroll
      for (int r = 0; r < 8; ++r)
        out[(size_t)(m0 + rt * 16 + rbase + r) * DIM + oc] = acc[rt * 4 + jt][r] * linv[r];
    }
  }
}

extern "C" void kernel_launch(void* const* d_in, const int* in_sizes, int n_in,
                              void* d_out, int out_size, void* d_ws, size_t ws_size,
                              hipStream_t stream) {
  (void)in_sizes; (void)n_in; (void)out_size; (void)ws_size;

  const float* z  = (const float*)d_in[0];
  const float* Wq = (const float*)d_in[1];
  const float* bq = (const float*)d_in[2];
  const float* Wk = (const float*)d_in[3];
  const float* bk = (const float*)d_in[4];
  const float* Wv = (const float*)d_in[5];
  const float* bv = (const float*)d_in[6];
  float* out = (float*)d_out;

  const size_t elems = (size_t)N_ASSETS * DIM;  // 4M elems
  __bf16* Qb = (__bf16*)d_ws;
  __bf16* Kb = Qb + elems;
  __bf16* Vt = Kb + elems;
  __bf16* zb = Vt + elems;
  __bf16* Wb = zb + elems;                      // 3 * 512*512 bf16

  // convert z + 3 weight matrices to bf16: (4M + 768K)/4 elems per thread-4
  const int cvt_threads = (N_ASSETS * DIM + 3 * DIM * DIM) / 4;
  CANN_cvt_kernel<<<cvt_threads / 256, 256, 0, stream>>>(z, Wq, Wk, Wv, zb, Wb);

  dim3 pgrid((N_ASSETS / 16) * (DIM / 16) / 8, 3, 1);
  CANN_proj_kernel<<<pgrid, 256, 0, stream>>>(zb, Wb, bq, bk, bv, Qb, Kb, Vt);

  CANN_attn_kernel<<<N_ASSETS / BR, 256, 0, stream>>>(Qb, Kb, Vt, out);
}
